// FeatureAttentionOverTime_40441412059865
// MI455X (gfx1250) — matrix operand order repaired and checked by reading, not compiled, not run on hardware
//
#include <hip/hip_runtime.h>

typedef __attribute__((ext_vector_type(16))) _Float16 v16h;
typedef __attribute__((ext_vector_type(8)))  float    v8f;
typedef __attribute__((ext_vector_type(2)))  _Float16 h2;

#define B_ 8
#define S_ 512
#define D_ 256
#define H_ 128

// Packed f16 tanh. CDNA5 has V_TANH_F16 (trans op). Prefer builtin if the
// toolchain exposes it; otherwise emit true16-syntax inline asm, writing the
// .l and .h halves of one 32-bit VGPR so the result is already packed.
__device__ __forceinline__ h2 tanh2(h2 x) {
#if __has_builtin(__builtin_amdgcn_tanh_f16)
    h2 r;
    r[0] = __builtin_amdgcn_tanh_f16(x[0]);
    r[1] = __builtin_amdgcn_tanh_f16(x[1]);
    return r;
#else
    unsigned xu = __builtin_bit_cast(unsigned, x);
    unsigned r;
    asm("v_tanh_f16 %0.l, %1.l" : "=v"(r) : "v"(xu));
    asm("v_tanh_f16 %0.h, %1.h" : "+v"(r) : "v"(xu));
    return __builtin_bit_cast(h2, r);
#endif
}

__global__ __launch_bounds__(256) void FeatureAttentionOverTime_kernel(
    const float* __restrict__ enc,   // (B,S,D)
    const float* __restrict__ dh,    // (B,H)
    const float* __restrict__ Wat,   // (H,H+1)
    const float* __restrict__ ba,    // (H,)
    const float* __restrict__ wv,    // (H,)
    float* __restrict__ out)         // (B,S,D)
{
    __shared__ __align__(16) float lds_dh[H_];
    __shared__ __align__(16) float lds_w0[H_];
    __shared__ __align__(16) float lds_dp[H_];
    __shared__ __align__(16) float lds_sc[8 * D_];   // 8 waves x 256 scores

    const int tid  = threadIdx.x;
    const int lane = tid & 31;
    const int wid  = tid >> 5;
    const int b    = blockIdx.x >> 6;                 // 64 blocks per batch
    const int s    = ((blockIdx.x & 63) << 3) + wid;  // one (b,s) row per wave

    // ---- stage dh row and w0 = W_attn[:,0] ----
    if (tid < H_) {
        lds_dh[tid] = dh[b * H_ + tid];
        lds_w0[tid] = Wat[tid * (H_ + 1)];
    }
    __syncthreads();

    // ---- dec_proj[b,:] = dh[b,:] @ Wd^T + b_attn  (tiny GEMV, per block) ----
    if (tid < H_) {
        float acc = ba[tid];
        const float* wrow = Wat + tid * (H_ + 1) + 1;
#pragma unroll 8
        for (int k = 0; k < H_; ++k) acc = fmaf(lds_dh[k], wrow[k], acc);
        lds_dp[tid] = acc;
    }
    __syncthreads();

    const int g8  = (lane >> 4) << 3;   // A-frag K group offset
    const int g16 = (lane >> 4) << 4;   // unused; B-frag also uses g8

    // ---- per-lane loop-invariant coefficients, packed f16 pairs ----
    // A-frag slot (chunk, v, h2): k = chunk*32 + (v>=4)*16 + g8 + (v&3)*2 + h2
    h2 w0pk[32], dppk[32];
#pragma unroll
    for (int i = 0; i < 32; ++i) {
        int ch = i >> 3, v = i & 7;
        int kb = ch * 32 + ((v >> 2) & 1) * 16 + g8 + (v & 3) * 2;
        float2 pw = *(const float2*)&lds_w0[kb];
        float2 pd = *(const float2*)&lds_dp[kb];
        w0pk[i][0] = (_Float16)pw.x; w0pk[i][1] = (_Float16)pw.y;
        dppk[i][0] = (_Float16)pd.x; dppk[i][1] = (_Float16)pd.y;
    }

    // ---- B fragment: w_v broadcast in all 16 columns ----
    // B-frag 32x16 f16: VGPR v holds K = (v>=4)*16 + g8 + (v&3)*2 (+0,+1)
    v16h bfr[4];
#pragma unroll
    for (int ch = 0; ch < 4; ++ch) {
#pragma unroll
        for (int v = 0; v < 8; ++v) {
            int k = ch * 32 + ((v >> 2) & 1) * 16 + g8 + (v & 3) * 2;
            float2 p = *(const float2*)&wv[k];
            bfr[ch][2 * v]     = (_Float16)p.x;
            bfr[ch][2 * v + 1] = (_Float16)p.y;
        }
    }

    const float* erow = enc + ((size_t)(b * S_ + s)) * D_;

    // ---- 16 tiles of 16 d-elements: energy -> tanh -> WMMA reduce over h ----
    for (int t = 0; t < 16; ++t) {
        float e = erow[t * 16 + (lane & 15)];       // row m = lane&15
        h2 e2; e2[0] = (_Float16)e; e2[1] = (_Float16)e;
        v8f c = {};
#pragma unroll
        for (int ch = 0; ch < 4; ++ch) {
            v16h a;
#pragma unroll
            for (int p = 0; p < 8; ++p) {
                h2 x = w0pk[ch * 8 + p] * e2 + dppk[ch * 8 + p];  // v_pk_fma_f16
                h2 xt = tanh2(x);                                  // 2x v_tanh_f16
                a[2 * p]     = xt[0];
                a[2 * p + 1] = xt[1];
            }
            c = __builtin_amdgcn_wmma_f32_16x16x32_f16(
                    false, a, false, bfr[ch], (short)0, c, false, false);
        }
        // D[m][n] identical for all n; lane 0 holds rows 0..7 (col 0),
        // lane 16 holds rows 8..15 (col 0)
        if ((lane & 15) == 0) {
            int rb = (lane >> 4) << 3;
#pragma unroll
            for (int j = 0; j < 8; ++j)
                lds_sc[wid * D_ + t * 16 + rb + j] = c[j];
        }
    }

    // ---- softmax over D=256 within this wave (same-wave LDS, DS in-order) ----
    float v[8];
#pragma unroll
    for (int j = 0; j < 8; ++j) v[j] = lds_sc[wid * D_ + lane + 32 * j];

    float m = v[0];
#pragma unroll
    for (int j = 1; j < 8; ++j) m = fmaxf(m, v[j]);
#pragma unroll
    for (int off = 16; off >= 1; off >>= 1) m = fmaxf(m, __shfl_xor(m, off));

    float sum = 0.0f;
#pragma unroll
    for (int j = 0; j < 8; ++j) {
        v[j] = __builtin_amdgcn_exp2f((v[j] - m) * 1.4426950408889634f);
        sum += v[j];
    }
#pragma unroll
    for (int off = 16; off >= 1; off >>= 1) sum += __shfl_xor(sum, off);

    float inv = 1.0f / sum;
    float* orow = out + ((size_t)(b * S_ + s)) * D_;
#pragma unroll
    for (int j = 0; j < 8; ++j) orow[lane + 32 * j] = v[j] * inv;
}

extern "C" void kernel_launch(void* const* d_in, const int* in_sizes, int n_in,
                              void* d_out, int out_size, void* d_ws, size_t ws_size,
                              hipStream_t stream) {
    (void)in_sizes; (void)n_in; (void)out_size; (void)d_ws; (void)ws_size;
    const float* enc = (const float*)d_in[0];
    const float* dh  = (const float*)d_in[1];
    const float* Wat = (const float*)d_in[2];
    const float* ba  = (const float*)d_in[3];
    const float* wv  = (const float*)d_in[4];
    float* out = (float*)d_out;

    dim3 grid(B_ * (S_ / 8));   // 512 blocks, 8 waves each -> one (b,s) per wave
    dim3 block(256);
    FeatureAttentionOverTime_kernel<<<grid, block, 0, stream>>>(enc, dh, Wat, ba, wv, out);
}
